// KALP_38285338476610
// MI455X (gfx1250) — compile-verified, matrix operand order checked by reading, not run on hardware
//
#include <hip/hip_runtime.h>
#include <hip/hip_bf16.h>
#include <math.h>

// ---------------------------------------------------------------------------
// GATv2 layer, N=2048 nodes, 128 in-features, 4 heads x 16 hidden.
//   k1: g_l = h@W_l, g_r = h@W_r          (WMMA f16 -> f32)
//   k2: e[i,j,h] -> masked softmax over j (fused in LDS, streams adj once)
//   k3: repack g_r into WMMA-B fragments  (f16)
//   k4: h' = A @ g_r per head, ELU        (WMMA f16 -> f32, prefetched)
// ---------------------------------------------------------------------------

typedef __attribute__((ext_vector_type(16))) _Float16 v16h;
typedef __attribute__((ext_vector_type(8)))  _Float16 v8h;
typedef __attribute__((ext_vector_type(8)))  float    v8f;

#define NN    2048
#define FIN   128
#define NH    4
#define FH    16          // hidden per head
#define FOUT  64          // NH*FH
#define KSTEP 32          // WMMA K for f16
#define NKAGG (NN / KSTEP)   // 64 K-steps in aggregation

// ---------------------------------------------------------------------------
// Kernel 1: projections. One wave per 16-row tile of h; computes the 16x64
// output tile for both W_l and W_r. A-fragments (rows of h) are hoisted and
// reused across both matrices and all four 16-col tiles.
// A 16x32 f16 fragment: lane L holds row M=L%16; K groups {0-7,16-23} for
// lanes 0-15 and {8-15,24-31} for lanes 16-31 (ISA 7.12.2).
// ---------------------------------------------------------------------------
__global__ __launch_bounds__(32) void gat_proj(const float* __restrict__ hin,
                                               const float* __restrict__ Wl,
                                               const float* __restrict__ Wr,
                                               float* __restrict__ gl,
                                               float* __restrict__ gr) {
  const int lane  = threadIdx.x;
  const int i0    = blockIdx.x * 16;
  const int mrow  = i0 + (lane & 15);
  const int khalf = (lane >> 4) * 8;

  // A fragments for K = 0..127 in four chunks of 32.
  v16h afrag[4];
#pragma unroll
  for (int kk = 0; kk < 4; ++kk) {
    const float* hp = hin + mrow * FIN + kk * KSTEP + khalf;
    const float4 g0 = *reinterpret_cast<const float4*>(hp + 0);
    const float4 g1 = *reinterpret_cast<const float4*>(hp + 4);
    const float4 g2 = *reinterpret_cast<const float4*>(hp + 16);
    const float4 g3 = *reinterpret_cast<const float4*>(hp + 20);
    const float tf[16] = {g0.x, g0.y, g0.z, g0.w, g1.x, g1.y, g1.z, g1.w,
                          g2.x, g2.y, g2.z, g2.w, g3.x, g3.y, g3.z, g3.w};
#pragma unroll
    for (int t = 0; t < 16; ++t) afrag[kk][t] = (_Float16)tf[t];
  }

  const int ncol  = lane & 15;
  const int mbase = i0 + ((lane >> 4) ? 8 : 0);

#pragma unroll
  for (int w = 0; w < 2; ++w) {
    const float* W = w ? Wr : Wl;
    float*       G = w ? gr : gl;
#pragma unroll
    for (int nt = 0; nt < 4; ++nt) {
      const int n = nt * 16 + ncol;
      v8f acc = {};
#pragma unroll
      for (int kk = 0; kk < 4; ++kk) {
        v16h b;
#pragma unroll
        for (int idx = 0; idx < 16; ++idx) {
          const int k = kk * KSTEP + ((idx & 8) ? 16 : 0) + khalf + (idx & 7);
          b[idx] = (_Float16)W[k * FOUT + n];
        }
        acc = __builtin_amdgcn_wmma_f32_16x16x32_f16(false, afrag[kk], false, b,
                                                     (short)0, acc, false, false);
      }
#pragma unroll
      for (int r = 0; r < 8; ++r)
        G[(size_t)(mbase + r) * FOUT + n] = acc[r];
    }
  }
}

// ---------------------------------------------------------------------------
// Kernel 2: per-row fused scores + masked softmax. One 256-thread block per
// node i; the full e-row (4 heads x 2048) lives in 32KB LDS, so adj is read
// exactly once and no f32 NxN tensor ever touches memory. g_r[i] and attn_w
// live in registers as float4; g_l rows stream via b128 loads (L2-resident).
// LeakyReLU computed as max(x, 0.2x). Output: f16 attention A[h][i][j].
// ---------------------------------------------------------------------------
__global__ __launch_bounds__(256) void gat_scores(const float* __restrict__ gl,
                                                  const float* __restrict__ gr,
                                                  const int* __restrict__ adj,
                                                  const float* __restrict__ attn_w,
                                                  _Float16* __restrict__ A) {
  __shared__ float e[NH][NN];
  __shared__ float red[8];
  __shared__ float bcast;

  const int i   = blockIdx.x;
  const int tid = threadIdx.x;

  // g_r[i] (64 floats) and attn_w (16 floats) in registers, vector loads.
  float4 gri4[16];
  float4 aw4[4];
  {
    const float4* grp = reinterpret_cast<const float4*>(gr + (size_t)i * FOUT);
    const float4* awp = reinterpret_cast<const float4*>(attn_w);
#pragma unroll
    for (int t = 0; t < 16; ++t) gri4[t] = grp[t];
#pragma unroll
    for (int t = 0; t < 4; ++t)  aw4[t] = awp[t];
  }

  const int* adjrow = adj + (size_t)i * NN;
  for (int j = tid; j < NN; j += 256) {
    const bool conn = adjrow[j] != 0;
    const float4* glj4 = reinterpret_cast<const float4*>(gl + (size_t)j * FOUT);
#pragma unroll
    for (int hh = 0; hh < NH; ++hh) {
      float s = 0.f;
#pragma unroll
      for (int q = 0; q < 4; ++q) {
        const float4 g = glj4[hh * 4 + q];
        const float4 r = gri4[hh * 4 + q];
        const float4 w = aw4[q];
        float x;
        x = r.x + g.x; x = fmaxf(x, 0.2f * x); s = fmaf(w.x, x, s);
        x = r.y + g.y; x = fmaxf(x, 0.2f * x); s = fmaf(w.y, x, s);
        x = r.z + g.z; x = fmaxf(x, 0.2f * x); s = fmaf(w.z, x, s);
        x = r.w + g.w; x = fmaxf(x, 0.2f * x); s = fmaf(w.w, x, s);
      }
      e[hh][j] = conn ? s : -__builtin_inff();
    }
  }
  __syncthreads();

  for (int hh = 0; hh < NH; ++hh) {
    // --- max over j ---
    float m = -__builtin_inff();
    for (int j = tid; j < NN; j += 256) m = fmaxf(m, e[hh][j]);
#pragma unroll
    for (int off = 16; off > 0; off >>= 1) m = fmaxf(m, __shfl_xor(m, off, 32));
    if ((tid & 31) == 0) red[tid >> 5] = m;
    __syncthreads();
    if (tid == 0) {
      float mm = red[0];
#pragma unroll
      for (int wv = 1; wv < 8; ++wv) mm = fmaxf(mm, red[wv]);
      bcast = mm;
    }
    __syncthreads();
    m = bcast;

    // --- sum of exp (overwrite e with exp) ---
    float s = 0.f;
    for (int j = tid; j < NN; j += 256) {
      const float ex = expf(e[hh][j] - m);
      e[hh][j] = ex;
      s += ex;
    }
#pragma unroll
    for (int off = 16; off > 0; off >>= 1) s += __shfl_xor(s, off, 32);
    __syncthreads();
    if ((tid & 31) == 0) red[tid >> 5] = s;
    __syncthreads();
    if (tid == 0) {
      float ss = 0.f;
#pragma unroll
      for (int wv = 0; wv < 8; ++wv) ss += red[wv];
      bcast = ss;
    }
    __syncthreads();
    const float inv = 1.0f / bcast;

    _Float16* Arow = A + ((size_t)hh * NN + i) * NN;
    for (int j = tid; j < NN; j += 256)
      Arow[j] = (_Float16)(e[hh][j] * inv);
    __syncthreads();
  }
}

// ---------------------------------------------------------------------------
// Kernel 3: repack g_r into WMMA B-fragments (f16). One wave per (head,
// k-step). B fragment: lane L holds column n=L%16, same K pattern as A.
// Stored so kernel 4 loads one contiguous 32B v16h per fragment.
// ---------------------------------------------------------------------------
__global__ __launch_bounds__(32) void gat_repack(const float* __restrict__ gr,
                                                 v16h* __restrict__ pb) {
  const int kk    = blockIdx.x;       // 0..63
  const int hh    = blockIdx.y;       // 0..3
  const int lane  = threadIdx.x;
  const int n     = lane & 15;
  const int khalf = (lane >> 4) * 8;
  v16h b;
#pragma unroll
  for (int idx = 0; idx < 16; ++idx) {
    const int k = kk * KSTEP + ((idx & 8) ? 16 : 0) + khalf + (idx & 7);
    b[idx] = (_Float16)gr[(size_t)k * FOUT + hh * FH + n];
  }
  pb[((size_t)hh * NKAGG + kk) * 32 + lane] = b;
}

// ---------------------------------------------------------------------------
// Kernel 4: aggregation h' = A @ g_r per head + ELU. Blocks of 4 waves; each
// wave owns one 16-row M-tile and all 4 waves share the same head's packed-B
// stream (L0 reuse). 64 chained wmma_f32_16x16x32_f16 per wave, with the A
// stream prefetched 8 K-steps ahead (global_prefetch_b8).
// ---------------------------------------------------------------------------
__global__ __launch_bounds__(128) void gat_agg(const _Float16* __restrict__ A,
                                               const v16h* __restrict__ pb,
                                               float* __restrict__ out) {
  const int wave  = threadIdx.x >> 5;              // 0..3
  const int lane  = threadIdx.x & 31;
  const int mt    = blockIdx.x * 4 + wave;         // 0..127
  const int hh    = blockIdx.y;                    // 0..3
  const int row   = mt * 16 + (lane & 15);
  const int khalf = (lane >> 4) * 8;

  const _Float16* Arow = A + ((size_t)hh * NN + row) * NN;
  const v16h*     pbh  = pb + (size_t)hh * NKAGG * 32 + lane;

  v8f acc = {};
  for (int kk = 0; kk < NKAGG; ++kk) {
    if (kk + 8 < NKAGG)                            // uniform branch: EXEC intact
      __builtin_prefetch(Arow + (kk + 8) * KSTEP, 0, 1);
    const v8h lo = *reinterpret_cast<const v8h*>(Arow + kk * KSTEP + khalf);
    const v8h hi = *reinterpret_cast<const v8h*>(Arow + kk * KSTEP + 16 + khalf);
    v16h a;
#pragma unroll
    for (int t = 0; t < 8; ++t) { a[t] = lo[t]; a[8 + t] = hi[t]; }
    const v16h b = pbh[(size_t)kk * 32];
    acc = __builtin_amdgcn_wmma_f32_16x16x32_f16(false, a, false, b,
                                                 (short)0, acc, false, false);
  }

  const int n     = lane & 15;
  const int mbase = mt * 16 + ((lane >> 4) ? 8 : 0);
#pragma unroll
  for (int r = 0; r < 8; ++r) {
    float v = acc[r];
    v = (v >= 0.f) ? v : expm1f(v);             // ELU(alpha=1)
    out[(size_t)(mbase + r) * FOUT + hh * FH + n] = v;
  }
}

// ---------------------------------------------------------------------------
extern "C" void kernel_launch(void* const* d_in, const int* in_sizes, int n_in,
                              void* d_out, int out_size, void* d_ws, size_t ws_size,
                              hipStream_t stream) {
  const float* h   = (const float*)d_in[0];   // [2048,128] f32
  const int*   adj = (const int*)  d_in[1];   // [2048,2048,1] i32
  const float* Wl  = (const float*)d_in[2];   // [128,64] f32
  const float* Wr  = (const float*)d_in[3];   // [128,64] f32
  const float* aw  = (const float*)d_in[4];   // [16] f32
  float* out = (float*)d_out;                 // [2048,64] f32

  // workspace layout (33.3 MB total)
  char* ws = (char*)d_ws;
  float*    gl = (float*)(ws);                                   // 512 KB
  float*    gr = (float*)(ws + (size_t)512 * 1024);              // 512 KB
  v16h*     pb = (v16h*) (ws + (size_t)1024 * 1024);             // 256 KB
  _Float16* A  = (_Float16*)(ws + (size_t)1024 * 1024 + 256 * 1024); // 32 MB

  gat_proj  <<<dim3(NN / 16),         32, 0, stream>>>(h, Wl, Wr, gl, gr);
  gat_scores<<<dim3(NN),             256, 0, stream>>>(gl, gr, adj, aw, A);
  gat_repack<<<dim3(NKAGG, NH),       32, 0, stream>>>(gr, pb);
  gat_agg   <<<dim3(NN / 64, NH),    128, 0, stream>>>(A, pb, out);
}
